// Encoder_49538152792843
// MI455X (gfx1250) — matrix operand order repaired and checked by reading, not compiled, not run on hardware
//
#include <hip/hip_runtime.h>
#include <cmath>

typedef __attribute__((ext_vector_type(8)))  __bf16 v8bf;
typedef __attribute__((ext_vector_type(16))) __bf16 v16bf;
typedef __attribute__((ext_vector_type(8)))  float  v8f;

#define DFEAT 512
#define TM 128
#define TN 64
#define TK 32
#define LDA 40   // padded bf16 stride (conflict-free, keeps 16B alignment: 80B rows)
#define LDB 40

union V16U { v16bf v; v8bf h[2]; };

__device__ __forceinline__ unsigned short f2bf(float f) {
  unsigned int u = __float_as_uint(f);
  unsigned int r = (u + 0x7FFFu + ((u >> 16) & 1u)) >> 16;  // round-to-nearest-even
  return (unsigned short)r;
}
__device__ __forceinline__ float bflo(unsigned int u) { return __uint_as_float(u << 16); }
__device__ __forceinline__ float bfhi(unsigned int u) { return __uint_as_float(u & 0xFFFF0000u); }

// hardware fire-and-forget f32 atomic add (no-return form, STOREcnt-tracked)
__device__ __forceinline__ void atomic_add_f32(float* p, float v) {
  unsigned long long a = (unsigned long long)p;
  asm volatile("global_atomic_add_f32 %0, %1, off" :: "v"(a), "v"(v) : "memory");
}

// ---- W [K][N] f32 -> Wt [N][K] bf16 (N-major: B fragments contiguous in K) ----
__global__ void wt_kernel(const float* __restrict__ W, unsigned short* __restrict__ Wt) {
  int idx = blockIdx.x * 256 + threadIdx.x;   // 0 .. 512*512-1
  int k = idx >> 9;
  int n = idx & 511;
  Wt[(size_t)n * DFEAT + k] = f2bf(W[(size_t)k * DFEAT + n]);
}

// ---- f32 -> packed bf16 (optionally fused ELU) ----
__global__ __launch_bounds__(256) void pack_bf16_kernel(const float* __restrict__ in,
                                                        unsigned short* __restrict__ out,
                                                        long long n4, int apply_elu) {
  long long i = (long long)blockIdx.x * 256 + threadIdx.x;
  if (i >= n4) return;
  float4 v = ((const float4*)in)[i];
  if (apply_elu) {
    v.x = v.x > 0.f ? v.x : expm1f(v.x);
    v.y = v.y > 0.f ? v.y : expm1f(v.y);
    v.z = v.z > 0.f ? v.z : expm1f(v.z);
    v.w = v.w > 0.f ? v.w : expm1f(v.w);
  }
  unsigned int u0 = (unsigned int)f2bf(v.x) | ((unsigned int)f2bf(v.y) << 16);
  unsigned int u1 = (unsigned int)f2bf(v.z) | ((unsigned int)f2bf(v.w) << 16);
  ((uint2*)out)[i] = make_uint2(u0, u1);
}

// ---- degree / norm ----
__global__ void deg_init_kernel(int* __restrict__ deg, int n) {
  int i = blockIdx.x * 256 + threadIdx.x;
  if (i < n) deg[i] = 1;                       // self loop
}
__global__ void deg_count_kernel(const int* __restrict__ dst, int* __restrict__ deg, int ne) {
  int i = blockIdx.x * 256 + threadIdx.x;
  if (i < ne) atomicAdd(&deg[dst[i]], 1);
}
__global__ void dinv_kernel(float* __restrict__ dinv, int n) {
  int i = blockIdx.x * 256 + threadIdx.x;
  if (i < n) {
    int d = ((const int*)dinv)[i];
    dinv[i] = rsqrtf((float)d);
  }
}

// ---- tiled bf16 WMMA GEMM: Hb[nrows x 512](bf16) = Xb(bf16) * W (Wt bf16 N-major) ----
__global__ __launch_bounds__(256) void gemm_kernel(const unsigned short* __restrict__ Xb,
                                                   const unsigned short* __restrict__ Wt,
                                                   unsigned short* __restrict__ Hb, int nrows) {
  __shared__ unsigned short sA[TM * LDA];
  __shared__ unsigned short sB[TN * LDB];

  const int tid  = threadIdx.x;
  const int m0   = blockIdx.x * TM;
  const int n0   = blockIdx.y * TN;
  const int w    = tid >> 5;
  const int wm   = w & 3;      // 4 M-slices of 32 rows
  const int wn   = w >> 2;     // 2 N-slices of 32 cols
  const int lane = tid & 31;
  const int lrow = lane & 15;
  const int hs   = lane >> 4;

  v8f acc00 = {}, acc01 = {}, acc10 = {}, acc11 = {};

  const int ldr = tid >> 2;         // 0..63
  const int ldc = (tid & 3) * 8;    // 0,8,16,24 (8 bf16 = 16B)
  const bool full = (m0 + TM) <= nrows;

  for (int kk = 0; kk < DFEAT; kk += TK) {
    // stage A: 128x32 bf16, straight 16B copies
    if (full) {
#pragma unroll
      for (int i = 0; i < 2; ++i) {
        int row = ldr + 64 * i;
        uint4 v = *(const uint4*)(Xb + (size_t)(m0 + row) * DFEAT + kk + ldc);
        *(uint4*)&sA[row * LDA + ldc] = v;
      }
    } else {
#pragma unroll
      for (int i = 0; i < 2; ++i) {
        int row = ldr + 64 * i;
        uint4 v = make_uint4(0u, 0u, 0u, 0u);
        if (m0 + row < nrows)
          v = *(const uint4*)(Xb + (size_t)(m0 + row) * DFEAT + kk + ldc);
        *(uint4*)&sA[row * LDA + ldc] = v;
      }
    }
    // stage B: 64x32 bf16
    {
      uint4 v = *(const uint4*)(Wt + (size_t)(n0 + ldr) * DFEAT + kk + ldc);
      *(uint4*)&sB[ldr * LDB + ldc] = v;
    }
    __syncthreads();

    // fragments per ISA 16-bit A(16x32): lanes0-15 K=0..7,16..23 ; lanes16-31 K=8..15,24..31
    V16U a0, a1, b0, b1;
    {
      int r0 = (wm * 32 + lrow) * LDA;
      a0.h[0] = *(const v8bf*)&sA[r0 + hs * 8];
      a0.h[1] = *(const v8bf*)&sA[r0 + 16 + hs * 8];
      int r1 = (wm * 32 + 16 + lrow) * LDA;
      a1.h[0] = *(const v8bf*)&sA[r1 + hs * 8];
      a1.h[1] = *(const v8bf*)&sA[r1 + 16 + hs * 8];
      int c0 = (wn * 32 + lrow) * LDB;
      b0.h[0] = *(const v8bf*)&sB[c0 + hs * 8];
      b0.h[1] = *(const v8bf*)&sB[c0 + 16 + hs * 8];
      int c1 = (wn * 32 + 16 + lrow) * LDB;
      b1.h[0] = *(const v8bf*)&sB[c1 + hs * 8];
      b1.h[1] = *(const v8bf*)&sB[c1 + 16 + hs * 8];
    }
    acc00 = __builtin_amdgcn_wmma_f32_16x16x32_bf16(false, a0.v, false, b0.v, (short)0, acc00, false, false);
    acc01 = __builtin_amdgcn_wmma_f32_16x16x32_bf16(false, a0.v, false, b1.v, (short)0, acc01, false, false);
    acc10 = __builtin_amdgcn_wmma_f32_16x16x32_bf16(false, a1.v, false, b0.v, (short)0, acc10, false, false);
    acc11 = __builtin_amdgcn_wmma_f32_16x16x32_bf16(false, a1.v, false, b1.v, (short)0, acc11, false, false);
    __syncthreads();
  }

  // epilogue: C/D layout lanes0-15 -> M=r, lanes16-31 -> M=8+r ; N=lrow ; store bf16
  const int gcol0 = n0 + wn * 32 + lrow;
  const int gcol1 = gcol0 + 16;
  const int growb = m0 + wm * 32 + hs * 8;
  if (full) {
#pragma unroll
    for (int r = 0; r < 8; ++r) {
      int g0 = growb + r;
      int g1 = growb + 16 + r;
      Hb[(size_t)g0 * DFEAT + gcol0] = f2bf(acc00[r]);
      Hb[(size_t)g0 * DFEAT + gcol1] = f2bf(acc01[r]);
      Hb[(size_t)g1 * DFEAT + gcol0] = f2bf(acc10[r]);
      Hb[(size_t)g1 * DFEAT + gcol1] = f2bf(acc11[r]);
    }
  } else {
#pragma unroll
    for (int r = 0; r < 8; ++r) {
      int g0 = growb + r;
      int g1 = growb + 16 + r;
      if (g0 < nrows) {
        Hb[(size_t)g0 * DFEAT + gcol0] = f2bf(acc00[r]);
        Hb[(size_t)g0 * DFEAT + gcol1] = f2bf(acc01[r]);
      }
      if (g1 < nrows) {
        Hb[(size_t)g1 * DFEAT + gcol0] = f2bf(acc10[r]);
        Hb[(size_t)g1 * DFEAT + gcol1] = f2bf(acc11[r]);
      }
    }
  }
}

// ---- out[i] = h[i]*dinv[i]^2 + bias (self-loop term + bias), h in bf16 ----
__global__ __launch_bounds__(256) void agg_init_kernel(const unsigned short* __restrict__ Hb,
                                                       const float* __restrict__ dinv,
                                                       const float* __restrict__ bias,
                                                       float* __restrict__ out, int n) {
  long long idx = (long long)blockIdx.x * 256 + threadIdx.x;
  int node = (int)(idx >> 7);
  if (node >= n) return;
  int c = (int)(idx & 127) * 4;
  float s = dinv[node];
  s *= s;
  uint2 hv = *(const uint2*)(Hb + (size_t)node * DFEAT + c);
  float4 bv = *(const float4*)(bias + c);
  float4 o;
  o.x = bflo(hv.x) * s + bv.x;
  o.y = bfhi(hv.x) * s + bv.y;
  o.z = bflo(hv.y) * s + bv.z;
  o.w = bfhi(hv.y) * s + bv.w;
  *(float4*)(out + (size_t)node * DFEAT + c) = o;
}

// ---- scatter-add: out[dst] += h[src]*dinv[src]*dinv[dst]; 64 threads/edge, 8 floats each ----
__global__ __launch_bounds__(256) void agg_edge_kernel(const unsigned short* __restrict__ Hb,
                                                       const int* __restrict__ src,
                                                       const int* __restrict__ dst,
                                                       const float* __restrict__ dinv,
                                                       float* __restrict__ out, int ne) {
  long long t = (long long)blockIdx.x * 256 + threadIdx.x;
  int e = (int)(t >> 6);
  if (e >= ne) return;
  e = __builtin_amdgcn_readfirstlane(e);   // wave-uniform -> scalar loads below
  int d  = ((int)(t & 63)) * 8;
  int s  = src[e];
  int dn = dst[e];
  float wgt = dinv[s] * dinv[dn];
  uint4 hv = *(const uint4*)(Hb + (size_t)s * DFEAT + d);   // 8 bf16 = 16B
  float* op = out + (size_t)dn * DFEAT + d;
  atomic_add_f32(op + 0, bflo(hv.x) * wgt);
  atomic_add_f32(op + 1, bfhi(hv.x) * wgt);
  atomic_add_f32(op + 2, bflo(hv.y) * wgt);
  atomic_add_f32(op + 3, bfhi(hv.y) * wgt);
  atomic_add_f32(op + 4, bflo(hv.z) * wgt);
  atomic_add_f32(op + 5, bfhi(hv.z) * wgt);
  atomic_add_f32(op + 6, bflo(hv.w) * wgt);
  atomic_add_f32(op + 7, bfhi(hv.w) * wgt);
}

// ---- final in-place ELU (f32) ----
__global__ __launch_bounds__(256) void elu_kernel(float* __restrict__ out, long long n4) {
  long long i = (long long)blockIdx.x * 256 + threadIdx.x;
  if (i >= n4) return;
  float4 v = ((float4*)out)[i];
  v.x = v.x > 0.f ? v.x : expm1f(v.x);
  v.y = v.y > 0.f ? v.y : expm1f(v.y);
  v.z = v.z > 0.f ? v.z : expm1f(v.z);
  v.w = v.w > 0.f ? v.w : expm1f(v.w);
  ((float4*)out)[i] = v;
}

extern "C" void kernel_launch(void* const* d_in, const int* in_sizes, int n_in,
                              void* d_out, int out_size, void* d_ws, size_t ws_size,
                              hipStream_t stream) {
  const float* x  = (const float*)d_in[0];
  const int*   ei = (const int*)d_in[1];
  const float* W1 = (const float*)d_in[2];
  const float* b1 = (const float*)d_in[3];
  const float* W2 = (const float*)d_in[4];
  const float* b2 = (const float*)d_in[5];
  float* out = (float*)d_out;

  const int n  = in_sizes[0] / DFEAT;   // 100000
  const int ne = in_sizes[1] / 2;       // 3200000
  const int* src = ei;
  const int* dst = ei + ne;

  // ws: h bf16 [n*512] | xb bf16 [n*512] | dinv f32 [n] | Wt1 bf16 | Wt2 bf16  (~206 MB)
  unsigned short* h  = (unsigned short*)d_ws;
  unsigned short* xb = h + (size_t)n * DFEAT;
  float* dinv = (float*)(xb + (size_t)n * DFEAT);
  unsigned short* wt1 = (unsigned short*)(dinv + n);
  unsigned short* wt2 = wt1 + DFEAT * DFEAT;

  long long nthr_node = (long long)n * (DFEAT / 4);
  int nb_node = (int)((nthr_node + 255) / 256);
  long long nthr_edge = (long long)ne * (DFEAT / 8);
  int nb_edge = (int)((nthr_edge + 255) / 256);
  dim3 gg((n + TM - 1) / TM, DFEAT / TN);

  wt_kernel<<<DFEAT * DFEAT / 256, 256, 0, stream>>>(W1, wt1);
  wt_kernel<<<DFEAT * DFEAT / 256, 256, 0, stream>>>(W2, wt2);
  pack_bf16_kernel<<<nb_node, 256, 0, stream>>>(x, xb, nthr_node, 0);

  deg_init_kernel<<<(n + 255) / 256, 256, 0, stream>>>((int*)dinv, n);
  deg_count_kernel<<<(ne + 255) / 256, 256, 0, stream>>>(dst, (int*)dinv, ne);
  dinv_kernel<<<(n + 255) / 256, 256, 0, stream>>>(dinv, n);

  // layer 1
  gemm_kernel<<<gg, 256, 0, stream>>>(xb, wt1, h, n);
  agg_init_kernel<<<nb_node, 256, 0, stream>>>(h, dinv, b1, out, n);
  agg_edge_kernel<<<nb_edge, 256, 0, stream>>>(h, src, dst, dinv, out, ne);
  pack_bf16_kernel<<<nb_node, 256, 0, stream>>>(out, xb, nthr_node, 1);  // ELU fused, feeds layer 2

  // layer 2
  gemm_kernel<<<gg, 256, 0, stream>>>(xb, wt2, h, n);
  agg_init_kernel<<<nb_node, 256, 0, stream>>>(h, dinv, b2, out, n);
  agg_edge_kernel<<<nb_edge, 256, 0, stream>>>(h, src, dst, dinv, out, ne);
  elu_kernel<<<nb_node, 256, 0, stream>>>(out, nthr_node);
}
